// ScaledDotProductAttention_21878563406413
// MI455X (gfx1250) — matrix operand order repaired and checked by reading, not compiled, not run on hardware
//
#include <hip/hip_runtime.h>
#include <math.h>

#define BATCH   4
#define SEQ     2048
#define DMODEL  1024
#define NEG_BIG -1.0e9f
#define SCALE   0.03125f   // 1/sqrt(1024)

#define BN   128           // keys per outer iteration
#define QPAD 1032          // Qs row stride in bf16 elems (516 dwords -> bank spread)
#define LSP  132           // Ls row stride in f32    (132 dwords  -> bank spread)
#define PSP  132           // Ps row stride in bf16   (66 dwords   -> bank spread, even)

typedef __attribute__((ext_vector_type(16))) __bf16        v16bf;
typedef __attribute__((ext_vector_type(8)))  float         v8f;
typedef __attribute__((ext_vector_type(8)))  unsigned int  v8u;

// fp32 -> bf16 bits, round-to-nearest-even
static __device__ __forceinline__ unsigned short f2bf(float f) {
    unsigned int u = __builtin_bit_cast(unsigned int, f);
    unsigned int r = u + 0x7FFFu + ((u >> 16) & 1u);
    return (unsigned short)(r >> 16);
}

// ---------------------------------------------------------------------------
// Pre-pass: K -> bf16 row-major [B][S][D]; V -> bf16 TRANSPOSED [B][D][S]
// ---------------------------------------------------------------------------
__global__ void cvt_kv_kernel(const float* __restrict__ K,
                              const float* __restrict__ V,
                              unsigned short* __restrict__ Kb,
                              unsigned short* __restrict__ Vt) {
    size_t idx = (size_t)blockIdx.x * blockDim.x + threadIdx.x;
    int d      = (int)(idx & (DMODEL - 1));
    size_t sd  = idx >> 10;
    int s      = (int)(sd & (SEQ - 1));
    int b      = (int)(sd >> 11);
    Kb[idx] = f2bf(K[idx]);
    Vt[((size_t)b * DMODEL + d) * SEQ + s] = f2bf(V[idx]);
}

// ---------------------------------------------------------------------------
// Fused flash attention: 1 workgroup (8 waves) = 16 query rows
// ---------------------------------------------------------------------------
__global__ void __launch_bounds__(256)
fa_kernel(const float* __restrict__ Q,
          const int*   __restrict__ Mask,
          const unsigned short* __restrict__ Kb,
          const unsigned short* __restrict__ Vt,
          float* __restrict__ Out) {
    __shared__ unsigned short Qs[16 * QPAD];   // Q tile, bf16, pre-scaled
    __shared__ float          Ls[16 * LSP];    // raw masked logits (16 x 128)
    __shared__ unsigned short Ps[16 * PSP];    // exp(logit - m) as bf16
    __shared__ float m_s[16], l_s[16], a_s[16];

    const int tid  = threadIdx.x;
    const int wave = tid >> 5;
    const int lane = tid & 31;
    const int ln   = lane & 15;
    const int g    = lane >> 4;          // lane half

    const int b  = blockIdx.x >> 7;      // SEQ/16 = 128 query tiles / batch
    const int qt = blockIdx.x & 127;
    const int m0 = qt << 4;

    // ---- load Q tile into LDS as bf16, pre-scaled by 1/sqrt(D)
    {
        const float* qbase = Q + ((size_t)b * SEQ + m0) * DMODEL;
        for (int i = tid; i < 16 * DMODEL; i += 256) {
            int r = i >> 10, d = i & (DMODEL - 1);
            Qs[r * QPAD + d] = f2bf(qbase[(size_t)r * DMODEL + d] * SCALE);
        }
        if (tid < 16) { m_s[tid] = -INFINITY; l_s[tid] = 0.0f; a_s[tid] = 0.0f; }
    }
    __syncthreads();

    // per-wave output accumulator: D-slice of 128 columns = 8 n-tiles of 16x16
    v8f acc[8];
#pragma unroll
    for (int t = 0; t < 8; ++t) acc[t] = (v8f){0.f,0.f,0.f,0.f,0.f,0.f,0.f,0.f};

    const int dbase = wave << 7;

    for (int kt = 0; kt < SEQ; kt += BN) {
        // ============ QK^T phase: wave w -> 16x16 logit slice ============
        {
            const int key = kt + (wave << 4) + ln;   // this lane's key column
            v8f c = (v8f){0.f,0.f,0.f,0.f,0.f,0.f,0.f,0.f};
            const unsigned short* kptr =
                Kb + ((size_t)b * SEQ + key) * DMODEL + (g << 4);
#pragma unroll 4
            for (int k0 = 0; k0 < DMODEL; k0 += 32) {
                // A fragment (16x32 bf16) gathered from Qs, contiguous k = 16*half + s
                v8u araw;
                const unsigned short* qrow = Qs + ln * QPAD + k0;
#pragma unroll
                for (int v = 0; v < 8; ++v) {
                    int kk = (g << 4) + (v << 1);
                    araw[v] = *(const unsigned int*)(qrow + kk);
                }
                // B fragment (32x16): lane = key column, 16 contiguous d
                v8u braw = *(const v8u*)(kptr + k0);
                c = __builtin_amdgcn_wmma_f32_16x16x32_bf16(
                        false, __builtin_bit_cast(v16bf, araw),
                        false, __builtin_bit_cast(v16bf, braw),
                        (short)0, c, false, false);
            }
            float madd = (Mask[(size_t)b * SEQ + key] != 0) ? NEG_BIG : 0.0f;
            int col = (wave << 4) + ln;
#pragma unroll
            for (int j = 0; j < 8; ++j) {
                int row = j + (g << 3);              // C layout: M = j (+8 hi half)
                Ls[row * LSP + col] = c[j] + madd;
            }
        }
        __syncthreads();

        // ============ online softmax stats: (wave,half) -> one row ============
        {
            int row = (wave << 1) + g;               // rows 0..15
            float mold = m_s[row];
            float lold = l_s[row];
            float xv[8];
            float mx = -INFINITY;
#pragma unroll
            for (int i = 0; i < 8; ++i) {
                xv[i] = Ls[row * LSP + ln + (i << 4)];
                mx = fmaxf(mx, xv[i]);
            }
#pragma unroll
            for (int off = 1; off < 16; off <<= 1)
                mx = fmaxf(mx, __shfl_xor(mx, off, 16));
            float mnew = fmaxf(mold, mx);
            float ssum = 0.0f;
#pragma unroll
            for (int i = 0; i < 8; ++i) {
                float p = __expf(xv[i] - mnew);
                Ps[row * PSP + ln + (i << 4)] = f2bf(p);
                ssum += p;
            }
#pragma unroll
            for (int off = 1; off < 16; off <<= 1)
                ssum += __shfl_xor(ssum, off, 16);
            if (ln == 0) {
                float al = __expf(mold - mnew);      // first iter: exp(-inf)=0
                m_s[row] = mnew;
                a_s[row] = al;
                l_s[row] = lold * al + ssum;
            }
        }
        __syncthreads();

        // ============ PV phase: wave w -> its 128-wide D slice ============
        {
            float al[8];
#pragma unroll
            for (int j = 0; j < 8; ++j) al[j] = a_s[j + (g << 3)];
#pragma unroll
            for (int t = 0; t < 8; ++t)
#pragma unroll
                for (int j = 0; j < 8; ++j) acc[t][j] *= al[j];

            // A fragments: P (16 x 128 bf16) as 4 chunks of K=32
            v16bf afr[4];
#pragma unroll
            for (int ch = 0; ch < 4; ++ch) {
                v8u araw;
                const unsigned short* prow = Ps + ln * PSP + (ch << 5);
#pragma unroll
                for (int v = 0; v < 8; ++v) {
                    int kk = (g << 4) + (v << 1);
                    araw[v] = *(const unsigned int*)(prow + kk);
                }
                afr[ch] = __builtin_bit_cast(v16bf, araw);
            }
#pragma unroll
            for (int t = 0; t < 8; ++t) {
                int d = dbase + (t << 4) + ln;       // lane = output d column
                const unsigned short* vptr =
                    Vt + ((size_t)b * DMODEL + d) * SEQ + kt + (g << 4);
#pragma unroll
                for (int ch = 0; ch < 4; ++ch) {
                    v8u braw = *(const v8u*)(vptr + (ch << 5));
                    acc[t] = __builtin_amdgcn_wmma_f32_16x16x32_bf16(
                                 false, afr[ch],
                                 false, __builtin_bit_cast(v16bf, braw),
                                 (short)0, acc[t], false, false);
                }
            }
        }
        __syncthreads();
    }

    // ---- normalize by 1/l and write out
    float rl[8];
#pragma unroll
    for (int j = 0; j < 8; ++j) rl[j] = 1.0f / l_s[j + (g << 3)];
    float* obase = Out + ((size_t)b * SEQ + m0) * DMODEL + dbase;
#pragma unroll
    for (int t = 0; t < 8; ++t) {
#pragma unroll
        for (int j = 0; j < 8; ++j) {
            int row = j + (g << 3);
            obase[(size_t)row * DMODEL + (t << 4) + ln] = acc[t][j] * rl[j];
        }
    }
}

// ---------------------------------------------------------------------------
extern "C" void kernel_launch(void* const* d_in, const int* in_sizes, int n_in,
                              void* d_out, int out_size, void* d_ws, size_t ws_size,
                              hipStream_t stream) {
    const float* q    = (const float*)d_in[0];
    const float* k    = (const float*)d_in[1];
    const float* v    = (const float*)d_in[2];
    const int*   mask = (const int*)d_in[3];

    const size_t total = (size_t)BATCH * SEQ * DMODEL;       // 8,388,608
    unsigned short* Kb = (unsigned short*)d_ws;              // 16.78 MB
    unsigned short* Vt = Kb + total;                         // 16.78 MB (transposed)

    cvt_kv_kernel<<<(int)(total / 256), 256, 0, stream>>>(k, v, Kb, Vt);

    fa_kernel<<<BATCH * (SEQ / 16), 256, 0, stream>>>(
        q, mask, Kb, Vt, (float*)d_out);
}
